// EMA_33045478376015
// MI455X (gfx1250) — compile-verified, matrix operand order checked by reading, not run on hardware
//
#include <hip/hip_runtime.h>
#include <hip/hip_bf16.h>
#include <math.h>

typedef __attribute__((ext_vector_type(16))) _Float16 v16h;
typedef __attribute__((ext_vector_type(8)))  _Float16 v8h;
typedef __attribute__((ext_vector_type(8)))  float    v8f;
typedef __attribute__((ext_vector_type(2)))  float    v2f;
typedef __attribute__((ext_vector_type(4)))  unsigned u32x4;
typedef __attribute__((ext_vector_type(8)))  unsigned u32x8;

#define BG   512
#define CG   32
#define HW   4096
#define GN_EPS 1e-3f

// ---- workspace layout (float offsets) ----
constexpr size_t N_HWPRE = (size_t)BG * 128 * CG;          // 2,097,152
constexpr size_t OFF_SIG   = N_HWPRE;                      // sigmoid(gemm) buf
constexpr size_t OFF_SUM   = OFF_SIG + N_HWPRE;
constexpr size_t OFF_SUMSQ = OFF_SUM   + (size_t)BG * CG;
constexpr size_t OFF_MU    = OFF_SUMSQ + (size_t)BG * CG;
constexpr size_t OFF_RS    = OFF_MU    + (size_t)BG * CG;
constexpr size_t OFF_MEAN2 = OFF_RS    + (size_t)BG * CG;
constexpr size_t OFF_X21   = OFF_MEAN2 + (size_t)BG * CG;
constexpr size_t OFF_X11   = OFF_X21   + (size_t)BG * CG;
constexpr size_t OFF_X2    = OFF_X11   + 64;               // conv out f32, 256 MB
constexpr size_t OFF_XH    = OFF_X2 + (size_t)BG * HW * CG;// f16 copy of x, 128 MB

__device__ __forceinline__ float sigmoidf_(float v) { return 1.0f / (1.0f + __expf(-v)); }

// global -> LDS async copy, 16B per lane (tracked by ASYNCcnt)
__device__ __forceinline__ void async_ld_b128(const void* gptr, const void* lptr) {
  unsigned loff = (unsigned)(uintptr_t)lptr;   // low 32 bits of generic = LDS byte offset
  asm volatile("global_load_async_to_lds_b128 %0, %1, off"
               :: "v"(loff), "v"(gptr) : "memory");
}
__device__ __forceinline__ void wait_asynccnt0() {
  asm volatile("s_wait_asynccnt 0x0" ::: "memory");
}

// ---------------------------------------------------------------------------
// K0: x (f32) -> xh (f16), 8 elements/thread, b128 stores
// ---------------------------------------------------------------------------
__global__ void k0_cvt(const float* __restrict__ x, _Float16* __restrict__ xh) {
  const size_t i = ((size_t)blockIdx.x * 256 + threadIdx.x) * 8;
  float4 a = *(const float4*)(x + i);
  float4 b = *(const float4*)(x + i + 4);
  v8h h;
  h[0]=(_Float16)a.x; h[1]=(_Float16)a.y; h[2]=(_Float16)a.z; h[3]=(_Float16)a.w;
  h[4]=(_Float16)b.x; h[5]=(_Float16)b.y; h[6]=(_Float16)b.z; h[7]=(_Float16)b.w;
  *(v8h*)(xh + i) = h;
}

// ---------------------------------------------------------------------------
// K1: per-group row means (x_h) and column means (x_w) -> hw_pre (bg,128,cg)
// ---------------------------------------------------------------------------
__global__ void k1_means(const float* __restrict__ x, float* __restrict__ ws) {
  __shared__ float accw[64 * 32];
  const int g = blockIdx.x, t = threadIdx.x;
  const int ch = t & 31, rg = t >> 5;
  for (int i = t; i < 64 * 32; i += 256) accw[i] = 0.f;
  __syncthreads();
  const float* gx = x + (size_t)g * HW * CG;
  float* hwp = ws + (size_t)g * 128 * CG;
  for (int r = rg; r < 64; r += 8) {
    float sh = 0.f;
    const float* row = gx + (size_t)r * 64 * CG + ch;
    for (int j = 0; j < 64; ++j) {
      float v = row[(size_t)j * CG];
      sh += v;
      atomicAdd(&accw[j * 32 + ch], v);           // ds_add_f32
    }
    hwp[r * CG + ch] = sh * (1.f / 64.f);
  }
  __syncthreads();
  for (int j = rg; j < 64; j += 8)
    hwp[(64 + j) * CG + ch] = accw[j * 32 + ch] * (1.f / 64.f);
}

// ---------------------------------------------------------------------------
// K2: hw_pre (bg*128, 32) x w1 (32,32) + b1, sigmoid  (V_WMMA_F32_16X16X4_F32)
// ---------------------------------------------------------------------------
__global__ void k2_gemm_sig(const float* __restrict__ hwp,
                            const float* __restrict__ w1,
                            const float* __restrict__ b1,
                            float* __restrict__ sig) {
  __shared__ float w1s[32 * 32];
  __shared__ float b1s[32];
  const int t = threadIdx.x;
  for (int i = t; i < 1024; i += 256) w1s[i] = w1[i];
  if (t < 32) b1s[t] = b1[t];
  __syncthreads();
  const int wv = t >> 5, lane = t & 31;
  const int hl = (lane < 16) ? 0 : 1;
  const int lm = lane & 15;
  const int mrow0 = (blockIdx.x * 8 + wv) * 16;
  v8f acc0 = {}, acc1 = {};
  for (int kk = 0; kk < 8; ++kk) {
    const int k0 = kk * 4 + hl * 2;
    v2f a;
    a.x = hwp[(size_t)(mrow0 + lm) * 32 + k0];
    a.y = hwp[(size_t)(mrow0 + lm) * 32 + k0 + 1];
    v2f bl, bh;
    bl.x = w1s[(k0)     * 32 + lm];      bl.y = w1s[(k0 + 1) * 32 + lm];
    bh.x = w1s[(k0)     * 32 + 16 + lm]; bh.y = w1s[(k0 + 1) * 32 + 16 + lm];
    acc0 = __builtin_amdgcn_wmma_f32_16x16x4_f32(false, a, false, bl, (short)0, acc0, false, false);
    acc1 = __builtin_amdgcn_wmma_f32_16x16x4_f32(false, a, false, bh, (short)0, acc1, false, false);
  }
  #pragma unroll
  for (int v = 0; v < 8; ++v) {
    const int m = v + hl * 8;
    sig[(size_t)(mrow0 + m) * 32 + lm]      = sigmoidf_(acc0[v] + b1s[lm]);
    sig[(size_t)(mrow0 + m) * 32 + 16 + lm] = sigmoidf_(acc1[v] + b1s[16 + lm]);
  }
}

// ---------------------------------------------------------------------------
// K3: gated = gx*sig_h*sig_w ; per-(g,c) sum & sumsq -> mu, rsqrt(var+eps)
// ---------------------------------------------------------------------------
__global__ void k3_stats(const float* __restrict__ x, float* __restrict__ ws) {
  __shared__ float ssum[32], ssq[32];
  const int g = blockIdx.x, t = threadIdx.x;
  const int ch = t & 31, pg = t >> 5;
  if (t < 32) { ssum[t] = 0.f; ssq[t] = 0.f; }
  __syncthreads();
  const float* gx = x + (size_t)g * HW * CG;
  const float* sg = ws + OFF_SIG + (size_t)g * 128 * CG;
  float s = 0.f, s2 = 0.f;
  for (int p = pg; p < HW; p += 8) {
    const int r = p >> 6, c = p & 63;
    float v = gx[(size_t)p * CG + ch] * sg[r * CG + ch] * sg[(64 + c) * CG + ch];
    s += v; s2 += v * v;
  }
  atomicAdd(&ssum[ch], s);
  atomicAdd(&ssq[ch], s2);
  __syncthreads();
  if (t < 32) {
    float mu  = ssum[t] * (1.f / HW);
    float var = ssq[t] * (1.f / HW) - mu * mu;
    ws[OFF_SUM   + g * CG + t] = ssum[t];
    ws[OFF_SUMSQ + g * CG + t] = ssq[t];
    ws[OFF_MU    + g * CG + t] = mu;
    ws[OFF_RS    + g * CG + t] = rsqrtf(var + GN_EPS);
  }
}

// ---------------------------------------------------------------------------
// K4: 3x3 SAME conv, implicit GEMM via V_WMMA_F32_16X16X32_F16.
//     Halo tile (18x66x32 f16) staged by the Tensor Data Mover:
//     one descriptor, iterate mode: 1 row (2048 elems) per iteration,
//     lds stride 66*32 elems, nrows iterations. Pads zeroed via ds_store_b128.
// ---------------------------------------------------------------------------
__global__ void k4_conv_wmma(const _Float16* __restrict__ xh,
                             const float* __restrict__ w3,
                             const float* __restrict__ b3,
                             float* __restrict__ x2) {
  extern __shared__ _Float16 smem[];
  _Float16* tileA = smem;                  // 18*66*32 halves
  _Float16* wB    = smem + 18 * 66 * 32;   // 9*2*32*16 halves
  const int g = blockIdx.y, rt = blockIdx.x, t = threadIdx.x;
  const int rowbase = rt * 16;

  // ---- zero pad columns (cc==0 / cc==65), disjoint from TDM region ----
  v8h zz = {};
  for (int j = t; j < 18 * 2 * 4; j += 256) {
    const int row = j >> 3, k = j & 7;
    const int cc = (k < 4) ? 0 : 65, q = k & 3;
    *(v8h*)(tileA + ((size_t)row * 66 + cc) * 32 + q * 8) = zz;
  }
  // ---- zero out-of-range halo rows ----
  if (rt == 0) for (int j = t; j < 264; j += 256) *(v8h*)(tileA + (size_t)j * 8) = zz;
  if (rt == 3) for (int j = t; j < 264; j += 256) *(v8h*)(tileA + (size_t)17 * 66 * 32 + (size_t)j * 8) = zz;

  // ---- stage weights in WMMA B-fragment order (tiny, VALU) ----
  for (int job = t; job < 9 * 2 * 32; job += 256) {
    const int tap = job / 64, L = job & 31;
    const int n = (((job >> 5) & 1) << 4) + (L & 15);
    const int kb0 = (L < 16) ? 0 : 8, kb1 = kb0 + 16;
    _Float16* dst = wB + job * 16;
    const float* wsrc = w3 + (size_t)tap * 32 * 32 + n;
    #pragma unroll
    for (int e = 0; e < 8; ++e) {
      dst[e]     = (_Float16)wsrc[(size_t)(kb0 + e) * 32];
      dst[8 + e] = (_Float16)wsrc[(size_t)(kb1 + e) * 32];
    }
  }

  // ---- TDM: one wave issues the tile DMA (EXEC ignored by TENSOR ops) ----
  if (t < 32) {
    const int fr    = (rowbase - 1 < 0) ? 0 : rowbase - 1;       // first valid row
    const int lastr = (rowbase + 16 > 63) ? 63 : rowbase + 16;
    const int nrows = lastr - fr + 1;                            // 17 or 18
    const int lrow0 = fr - (rowbase - 1);                        // dest tile row
    const unsigned long long gaddr =
        (unsigned long long)(uintptr_t)(xh + (size_t)g * HW * CG + (size_t)fr * 2048);
    const unsigned ldsb =
        (unsigned)(uintptr_t)(tileA + ((size_t)lrow0 * 66 + 1) * 32);
    // D# group0: count=1 | lds_addr | global_addr(57b) | type=2
    u32x4 g0 = { 1u, ldsb, (unsigned)gaddr,
                 (unsigned)((gaddr >> 32) & 0x01FFFFFFull) | 0x80000000u };
    // D# group1: data_size=2B(code1), iterate_enable, dims/strides (elem units)
    u32x8 g1 = { (1u << 16) | (1u << 19),          // data_size | iterate_enable
                 (2048u << 16),                    // tensor_dim0 lo
                 (64u << 16),                      // tensor_dim0 hi=0 | tensor_dim1 lo
                 (2048u << 16),                    // tensor_dim1 hi=0 | tile_dim0=2048
                 1u,                               // tile_dim1=1 | tile_dim2=0
                 2048u,                            // tensor_dim0_stride lo
                 (2048u << 16),                    // t0s hi=0 | tensor_dim1_stride lo
                 0u };
    // D# group2 (iterate): tensor_dim2=1 | lds_addr_increment | ga_increment | count
    u32x4 g2 = { 1u, 66u * 32u, 2048u, ((unsigned)(nrows - 1)) << 16 };
    u32x4 g3 = { 0u, 0u, 0u, 0u };
    asm volatile("tensor_load_to_lds %0, %1, %2, %3"
                 :: "s"(g0), "s"(g1), "s"(g2), "s"(g3) : "memory");
    __builtin_amdgcn_s_wait_tensorcnt(0);
  }
  __syncthreads();

  const int wv = t >> 5, lane = t & 31;
  const int hl = (lane < 16) ? 0 : 1;
  const int lm = lane & 15;
  const int kb0 = hl * 8;
  const float bias0 = b3[lm], bias1 = b3[16 + lm];

  for (int mt = wv; mt < 64; mt += 8) {
    const int lrow = mt >> 2;
    const int cbase = (mt & 3) * 16;
    v8f acc0 = {}, acc1 = {};
    #pragma unroll
    for (int tap = 0; tap < 9; ++tap) {
      const int dy = tap / 3 - 1, dx = tap % 3 - 1;
      const int tr = lrow + 1 + dy;
      const _Float16* ap = tileA + ((size_t)tr * 66 + (cbase + lm + dx + 1)) * 32;
      v8h alo = *(const v8h*)(ap + kb0);
      v8h ahi = *(const v8h*)(ap + kb0 + 16);
      v16h a = __builtin_shufflevector(alo, ahi, 0,1,2,3,4,5,6,7,8,9,10,11,12,13,14,15);
      const _Float16* b0p = wB + ((tap * 2 + 0) * 32 + lane) * 16;
      const _Float16* b1p = wB + ((tap * 2 + 1) * 32 + lane) * 16;
      v8h b0l = *(const v8h*)(b0p),  b0h = *(const v8h*)(b0p + 8);
      v8h b1l = *(const v8h*)(b1p),  b1h = *(const v8h*)(b1p + 8);
      v16h b0 = __builtin_shufflevector(b0l, b0h, 0,1,2,3,4,5,6,7,8,9,10,11,12,13,14,15);
      v16h b1 = __builtin_shufflevector(b1l, b1h, 0,1,2,3,4,5,6,7,8,9,10,11,12,13,14,15);
      acc0 = __builtin_amdgcn_wmma_f32_16x16x32_f16(false, a, false, b0, (short)0, acc0, false, false);
      acc1 = __builtin_amdgcn_wmma_f32_16x16x32_f16(false, a, false, b1, (short)0, acc1, false, false);
    }
    float* orow = x2 + ((size_t)g * HW + (size_t)(rowbase + lrow) * 64) * CG;
    #pragma unroll
    for (int v = 0; v < 8; ++v) {
      const int pcol = cbase + v + hl * 8;
      orow[(size_t)pcol * CG + lm]      = acc0[v] + bias0;
      orow[(size_t)pcol * CG + 16 + lm] = acc1[v] + bias1;
    }
  }
}

// ---------------------------------------------------------------------------
// K4b: channel means of x2 -> softmax x21 ; x11 = softmax(beta) (exact)
// ---------------------------------------------------------------------------
__global__ void k4b_reduce(const float* __restrict__ x2,
                           const float* __restrict__ gnbeta,
                           float* __restrict__ ws) {
  __shared__ float ssum[32];
  const int g = blockIdx.x, t = threadIdx.x;
  const int ch = t & 31, pg = t >> 5;
  if (t < 32) ssum[t] = 0.f;
  __syncthreads();
  const float* xx = x2 + (size_t)g * HW * CG;
  float s = 0.f;
  for (int p = pg; p < HW; p += 8) {
    __builtin_prefetch(xx + (size_t)(p + 64) * CG + ch, 0, 0);   // global_prefetch_b8
    s += xx[(size_t)p * CG + ch];
  }
  atomicAdd(&ssum[ch], s);
  __syncthreads();
  if (t == 0) {
    float e[32], m = -1e30f;
    for (int c = 0; c < 32; ++c) { e[c] = ssum[c] * (1.f / HW); m = fmaxf(m, e[c]); }
    float den = 0.f;
    for (int c = 0; c < 32; ++c) { e[c] = __expf(e[c] - m); den += e[c]; }
    for (int c = 0; c < 32; ++c) {
      ws[OFF_MEAN2 + g * CG + c] = ssum[c] * (1.f / HW);
      ws[OFF_X21   + g * CG + c] = e[c] / den;
    }
    if (g == 0) {
      float mb = -1e30f, d2 = 0.f, eb[32];
      for (int c = 0; c < 32; ++c) mb = fmaxf(mb, gnbeta[c]);
      for (int c = 0; c < 32; ++c) { eb[c] = __expf(gnbeta[c] - mb); d2 += eb[c]; }
      for (int c = 0; c < 32; ++c) ws[OFF_X11 + c] = eb[c] / d2;
    }
  }
}

// ---------------------------------------------------------------------------
// K5: flat-reshape weighted sums -> sigmoid gate -> output
//     sigmoid table staged via async global->LDS b128 copies
// ---------------------------------------------------------------------------
__global__ void k5_final(const float* __restrict__ x,
                         const float* __restrict__ gamma,
                         const float* __restrict__ beta,
                         const float* __restrict__ ws,
                         float* __restrict__ out) {
  __shared__ alignas(16) float sgs[128 * 32];
  __shared__ float x11s[32], x21s[32], mus[32], rss[32], gms[32], bts[32];
  const int g = blockIdx.x, t = threadIdx.x;
  const float* sg = ws + OFF_SIG + (size_t)g * 128 * CG;
  for (int i = t; i < 1024; i += 256)          // 1024 x 16B async chunks
    async_ld_b128(sg + (size_t)i * 4, sgs + (size_t)i * 4);
  if (t < 32) {
    x11s[t] = ws[OFF_X11 + t];
    x21s[t] = ws[OFF_X21 + g * CG + t];
    mus[t]  = ws[OFF_MU  + g * CG + t];
    rss[t]  = ws[OFF_RS  + g * CG + t];
    gms[t]  = gamma[t];  bts[t] = beta[t];
  }
  wait_asynccnt0();
  __syncthreads();
  const float* gx  = x  + (size_t)g * HW * CG;
  const float* x2g = ws + OFF_X2 + (size_t)g * HW * CG;   // flat (cg, hw) reinterpret
  float* og = out + (size_t)g * HW * CG;
  for (int p = t; p < HW; p += 256) {
    float wsum = 0.f;
    #pragma unroll 4
    for (int c = 0; c < 32; ++c) {
      const size_t i = (size_t)c * HW + p;                // flat idx within group
      const int r2 = (int)(i >> 11), c2 = (int)((i >> 5) & 63), ch2 = (int)(i & 31);
      const float x2v = x2g[i];
      const float gxv = gx[i];
      const float gated = gxv * sgs[r2 * 32 + ch2] * sgs[(64 + c2) * 32 + ch2];
      const float x1v = (gated - mus[ch2]) * rss[ch2] * gms[ch2] + bts[ch2];
      wsum += x11s[c] * x2v + x21s[c] * x1v;
    }
    const float wsig = sigmoidf_(wsum);
    const int r = p >> 6, cl = p & 63;
    const float* grow = gx + ((size_t)r * 64 + cl) * CG;
    float* orow = og + ((size_t)r * 64 + cl) * CG;
    #pragma unroll
    for (int ch = 0; ch < 32; ++ch) orow[ch] = grow[ch] * wsig;
  }
}

// ---------------------------------------------------------------------------
extern "C" void kernel_launch(void* const* d_in, const int* in_sizes, int n_in,
                              void* d_out, int out_size, void* d_ws, size_t ws_size,
                              hipStream_t stream) {
  (void)in_sizes; (void)n_in; (void)out_size; (void)ws_size;
  const float* x  = (const float*)d_in[0];
  const float* w1 = (const float*)d_in[1];
  const float* b1 = (const float*)d_in[2];
  const float* w3 = (const float*)d_in[3];
  const float* b3 = (const float*)d_in[4];
  const float* gn_gamma = (const float*)d_in[5];
  const float* gn_beta  = (const float*)d_in[6];
  float* ws  = (float*)d_ws;
  float* out = (float*)d_out;
  _Float16* xh = (_Float16*)(ws + OFF_XH);

  k0_cvt     <<<32768, 256, 0, stream>>>(x, xh);          // 67.1M elems / 8 per thread
  k1_means   <<<BG,    256, 0, stream>>>(x, ws);
  k2_gemm_sig<<<512,   256, 0, stream>>>(ws, w1, b1, ws + OFF_SIG);
  k3_stats   <<<BG,    256, 0, stream>>>(x, ws);
  const size_t ldsA = (size_t)(18 * 66 * 32 + 9 * 2 * 32 * 16) * sizeof(_Float16); // ~94.5 KB
  k4_conv_wmma<<<dim3(4, BG), 256, ldsA, stream>>>(xh, w3, b3, ws + OFF_X2);
  k4b_reduce <<<BG,    256, 0, stream>>>(ws + OFF_X2, gn_beta, ws);
  k5_final   <<<BG,    256, 0, stream>>>(x, gn_gamma, gn_beta, ws, out);
}
